// AnimeStyleAttention_76106820485270
// MI455X (gfx1250) — matrix operand order repaired and checked by reading, not compiled, not run on hardware
//
#include <hip/hip_runtime.h>
#include <hip/hip_bf16.h>
#include <math.h>

// ---------------------------------------------------------------------------
// AnimeStyleAttention on MI455X (gfx1250): bf16 WMMA flash attention pipeline
//   B=4, N=2048, C=512, H=8, Dh=64
// v4: fix async-builtin pointer types (toolchain wants v4i* params).
// ---------------------------------------------------------------------------

#define DIM     512
#define NHEADS  8
#define HDIM    64
#define SEQ     2048
#define BATCH   4
#define MROWS   (BATCH * SEQ)   // 8192
#define KVSTEP  32
#define KVITERS (SEQ / KVSTEP)  // 64

typedef __attribute__((ext_vector_type(16))) __bf16 v16bf;
typedef __attribute__((ext_vector_type(8)))  __bf16 v8bf;
typedef __attribute__((ext_vector_type(8)))  float  v8f;
typedef int v4i_ __attribute__((vector_size(16)));   // builtin's pointee type

// ---- CDNA5 async global->LDS copy (ASYNCcnt-tracked), with safe fallback --
#if defined(__has_builtin)
#if __has_builtin(__builtin_amdgcn_global_load_async_to_lds_b128)
#define HAVE_ASYNC_LDS 1
#endif
#endif

#ifdef HAVE_ASYNC_LDS
__device__ __forceinline__ void cp16_g2l(const __bf16* g, __bf16* l) {
    __builtin_amdgcn_global_load_async_to_lds_b128(
        (v4i_*)g, (v4i_*)l, 0, 0);
}
__device__ __forceinline__ void wait_async_copies() {
#if __has_builtin(__builtin_amdgcn_s_wait_asynccnt)
    __builtin_amdgcn_s_wait_asynccnt(0);
#else
    asm volatile("s_wait_asynccnt 0" ::: "memory");
#endif
}
#else
__device__ __forceinline__ void cp16_g2l(const __bf16* g, __bf16* l) {
    *(v8bf*)l = *(const v8bf*)g;          // global_load_b128 + ds_store_b128
}
__device__ __forceinline__ void wait_async_copies() {}
#endif

// ---- WMMA fragment loaders (CDNA5 16x16x32 bf16 layouts, wave32) ----------

// A (16xK slice, K=32): lane<16 holds K {0..7,16..23}, lane>=16 holds
// K {8..15,24..31}, row = lane&15.   src: bf16 row-major, ld elements.
__device__ __forceinline__ v16bf load_a_bf16(const __bf16* p, int ld) {
    const int lane = threadIdx.x & 31;
    const int row  = lane & 15;
    const int ko   = (lane >> 4) << 3;
    const __bf16* r = p + row * ld + ko;
    v8bf lo = *(const v8bf*)(r);
    v8bf hi = *(const v8bf*)(r + 16);
    v16bf a;
#pragma unroll
    for (int i = 0; i < 8; ++i) { a[i] = lo[i]; a[i + 8] = hi[i]; }
    return a;
}

// B (32x16 tile): col = lane&15; lane<16 holds K {0..7,16..23}, lane>=16 holds K {8..15,24..31}.
__device__ __forceinline__ v16bf load_b_bf16(const __bf16* p, int ld) {
    const int lane = threadIdx.x & 31;
    const int col  = lane & 15;
    const int ko   = (lane >> 4) << 3;
    v16bf b;
#pragma unroll
    for (int i = 0; i < 8; ++i) {
        b[i]     = p[(ko + i) * ld + col];
        b[i + 8] = p[(ko + 16 + i) * ld + col];
    }
    return b;
}

#define WMMA_BF16(a, b, c) \
    __builtin_amdgcn_wmma_f32_16x16x32_bf16(false, (a), false, (b), (short)0, (c), false, false)

// ---------------------------------------------------------------------------
// Kernel 0: elementwise f32 -> bf16 conversion (x and weights, done once)
// ---------------------------------------------------------------------------
__global__ __launch_bounds__(256) void cvt_f32_bf16_kernel(
    const float* __restrict__ src, __bf16* __restrict__ dst)
{
    const int i = (blockIdx.x * 256 + threadIdx.x) * 8;
    v8f s = *(const v8f*)(src + i);
    v8bf d;
#pragma unroll
    for (int j = 0; j < 8; ++j) d[j] = (__bf16)s[j];
    *(v8bf*)(dst + i) = d;
}

// ---------------------------------------------------------------------------
// Kernel 1: style gating MLP  gate = sigmoid(gelu(style@Ws1+bs1)@Ws2+bs2)
// ---------------------------------------------------------------------------
__global__ __launch_bounds__(DIM) void gate_mlp_kernel(
    const float* __restrict__ style,
    const float* __restrict__ Ws1, const float* __restrict__ bs1,
    const float* __restrict__ Ws2, const float* __restrict__ bs2,
    float* __restrict__ gate)
{
    __shared__ float s_in[DIM];
    __shared__ float s_h1[DIM];
    const int c = threadIdx.x;
    const int b = blockIdx.x;
    s_in[c] = style[b * DIM + c];
    __syncthreads();
    float acc = bs1[c];
#pragma unroll 8
    for (int k = 0; k < DIM; ++k) acc = fmaf(s_in[k], Ws1[k * DIM + c], acc);
    s_h1[c] = 0.5f * acc * (1.0f + erff(acc * 0.70710678118654752f));  // exact GELU
    __syncthreads();
    float acc2 = bs2[c];
#pragma unroll 8
    for (int k = 0; k < DIM; ++k) acc2 = fmaf(s_h1[k], Ws2[k * DIM + c], acc2);
    gate[b * DIM + c] = 1.0f / (1.0f + __expf(-acc2));
}

// ---------------------------------------------------------------------------
// Kernel 2: QKV projection (bf16 x bf16 -> bf16). grid.z picks {q,k,v}.
// Each wave: 32x64 output tile (2 A-frags x 4 B-frags -> 8 WMMA / K-step).
// k stored transposed per head: kT[(b*H+h)*64 + d][seq]; q,v row-major.
// ---------------------------------------------------------------------------
__global__ __launch_bounds__(256) void qkv_proj_kernel(
    const __bf16* __restrict__ xb,
    const __bf16* __restrict__ Wq, const float* __restrict__ bq,
    const __bf16* __restrict__ Wk, const float* __restrict__ bk,
    const __bf16* __restrict__ Wv, const float* __restrict__ bv,
    __bf16* __restrict__ qout, __bf16* __restrict__ kT, __bf16* __restrict__ vout)
{
    const int which = blockIdx.z;
    const __bf16* W   = (which == 0) ? Wq : (which == 1) ? Wk : Wv;
    const float* bias = (which == 0) ? bq : (which == 1) ? bk : bv;

    const int wave = threadIdx.x >> 5;
    const int lane = threadIdx.x & 31;
    const int row0 = blockIdx.x * 256 + wave * 32;   // [0, 8192)
    const int col0 = blockIdx.y * 64;

    v8f acc[2][4] = {};
    const __bf16* ap = xb + (size_t)row0 * DIM;
#pragma unroll 2
    for (int k0 = 0; k0 < DIM; k0 += 32) {
        v16bf a0 = load_a_bf16(ap + k0, DIM);
        v16bf a1 = load_a_bf16(ap + (size_t)16 * DIM + k0, DIM);
#pragma unroll
        for (int j = 0; j < 4; ++j) {
            v16bf bfr = load_b_bf16(W + (size_t)k0 * DIM + col0 + j * 16, DIM);
            acc[0][j] = WMMA_BF16(a0, bfr, acc[0][j]);
            acc[1][j] = WMMA_BF16(a1, bfr, acc[1][j]);
        }
    }

    const int g = lane >> 4;   // row-half select
    const int n = lane & 15;   // column within tile
    if (which == 1) {
        // k: transposed per-head store -> contiguous B-fragments in attention
#pragma unroll
        for (int m = 0; m < 2; ++m) {
#pragma unroll
            for (int j = 0; j < 4; ++j) {
                const int c  = col0 + j * 16 + n;
                const float bb = bias[c];
                const int h = c >> 6, d = c & 63;
#pragma unroll
                for (int r = 0; r < 8; ++r) {
                    const int rowg = row0 + m * 16 + g * 8 + r;
                    const int b_ = rowg >> 11, seq = rowg & (SEQ - 1);
                    kT[((size_t)(b_ * NHEADS + h) * HDIM + d) * SEQ + seq] =
                        (__bf16)(acc[m][j][r] + bb);
                }
            }
        }
    } else {
        __bf16* out = (which == 0) ? qout : vout;
#pragma unroll
        for (int m = 0; m < 2; ++m) {
#pragma unroll
            for (int j = 0; j < 4; ++j) {
                const int c  = col0 + j * 16 + n;
                const float bb = bias[c];
#pragma unroll
                for (int r = 0; r < 8; ++r) {
                    const int rowg = row0 + m * 16 + g * 8 + r;
                    out[(size_t)rowg * DIM + c] = (__bf16)(acc[m][j][r] + bb);
                }
            }
        }
    }
}

// ---------------------------------------------------------------------------
// Kernel 3: flash attention + gate.  Block = 128 query rows of one (b,h),
// 8 waves x 16 rows.  K/V tiles (32 kv positions) double-buffered in LDS via
// async global->LDS copies; online softmax with 16-lane xor reductions;
// P round-trips through wave-private LDS to reach A-fragment layout.
// ---------------------------------------------------------------------------
__global__ __launch_bounds__(256) void flash_attn_kernel(
    const __bf16* __restrict__ q, const __bf16* __restrict__ kT,
    const __bf16* __restrict__ v, const float* __restrict__ gate,
    __bf16* __restrict__ ao)
{
    __shared__ __bf16 kTs[2][HDIM * KVSTEP];   // [dh][kv]  row stride 32  (4KB/buf)
    __shared__ __bf16 vs [2][KVSTEP * HDIM];   // [kv][dh]  row stride 64  (4KB/buf)
    __shared__ __bf16 pbuf[8][16 * KVSTEP];    // per-wave 16x32 P tile     (8KB)

    const int tid  = threadIdx.x;
    const int wave = tid >> 5;
    const int lane = tid & 31;
    const int h  = blockIdx.y;
    const int b  = blockIdx.z;
    const int q0 = blockIdx.x * 128 + wave * 16;
    const int bh = b * NHEADS + h;

    const __bf16* ktbase = kT + (size_t)bh * HDIM * SEQ;                // [64][2048]
    const __bf16* vbase  = v + (size_t)b * SEQ * DIM + h * HDIM;        // rows: DIM

    // staging assignment: 256 threads x 16B per tile
    const int krow = tid >> 2, kchunk = (tid & 3) * 8;   // kT: 64 rows x 4 chunks
    const int vrow = tid >> 3, vchunk = (tid & 7) * 8;   // V : 32 rows x 8 chunks

    // Q fragments (16 x 64 = two K=32 A-fragments), loaded once
    const __bf16* qp = q + (size_t)(b * SEQ + q0) * DIM + h * HDIM;
    const v16bf qa0 = load_a_bf16(qp, DIM);
    const v16bf qa1 = load_a_bf16(qp + 32, DIM);

    v8f acc[4] = {};
    float mrow[8], lrow[8];
#pragma unroll
    for (int r = 0; r < 8; ++r) { mrow[r] = -1e30f; lrow[r] = 0.0f; }

    __bf16* pw = pbuf[wave];
    const float scale = 0.125f;           // 1/sqrt(Dh)

    // prologue: stage tile 0
    cp16_g2l(ktbase + (size_t)krow * SEQ + kchunk, &kTs[0][krow * KVSTEP + kchunk]);
    cp16_g2l(vbase + (size_t)vrow * DIM + vchunk,  &vs[0][vrow * HDIM + vchunk]);
    wait_async_copies();
    __syncthreads();

    for (int it = 0; it < KVITERS; ++it) {
        const int cur = it & 1;
        const int kv0 = it * KVSTEP;

        // stage next tile into the other buffer (overlaps with compute)
        if (it + 1 < KVITERS) {
            const int kvn = kv0 + KVSTEP;
            cp16_g2l(ktbase + (size_t)krow * SEQ + kvn + kchunk,
                     &kTs[cur ^ 1][krow * KVSTEP + kchunk]);
            cp16_g2l(vbase + (size_t)(kvn + vrow) * DIM + vchunk,
                     &vs[cur ^ 1][vrow * HDIM + vchunk]);
        }
        if (it + 2 < KVITERS) {   // keep the tile after next warm in L2
            __builtin_prefetch(ktbase + kv0 + 2 * KVSTEP, 0, 1);
            __builtin_prefetch(vbase + (size_t)(kv0 + 2 * KVSTEP) * DIM, 0, 1);
        }

        // ---- scores S[16x32] = Q (16x64) x K^T (64x32), from LDS ---------
        v8f s0 = {}, s1 = {};
        {
            v16bf b00 = load_b_bf16(&kTs[cur][0],              KVSTEP);  // dh 0..31,  kv 0..15
            v16bf b01 = load_b_bf16(&kTs[cur][32 * KVSTEP],    KVSTEP);  // dh 32..63, kv 0..15
            s0 = WMMA_BF16(qa0, b00, s0);
            s0 = WMMA_BF16(qa1, b01, s0);
            v16bf b10 = load_b_bf16(&kTs[cur][16],             KVSTEP);  // kv 16..31
            v16bf b11 = load_b_bf16(&kTs[cur][32 * KVSTEP + 16], KVSTEP);
            s1 = WMMA_BF16(qa0, b10, s1);
            s1 = WMMA_BF16(qa1, b11, s1);
        }

        // ---- online softmax (row = r + 8*(lane>>4), cols across 16 lanes)
        float mnew[8], fsc[8];
#pragma unroll
        for (int r = 0; r < 8; ++r) {
            float mx = fmaxf(s0[r], s1[r]) * scale;
#pragma unroll
            for (int off = 1; off < 16; off <<= 1)
                mx = fmaxf(mx, __shfl_xor(mx, off, 32));
            mnew[r] = fmaxf(mrow[r], mx);
            fsc[r]  = __expf(mrow[r] - mnew[r]);
            mrow[r] = mnew[r];
        }
#pragma unroll
        for (int r = 0; r < 8; ++r) {
            float p0 = __expf(s0[r] * scale - mnew[r]);
            float p1 = __expf(s1[r] * scale - mnew[r]);
            s0[r] = p0; s1[r] = p1;
            float rs = p0 + p1;
#pragma unroll
            for (int off = 1; off < 16; off <<= 1)
                rs += __shfl_xor(rs, off, 32);
            lrow[r] = lrow[r] * fsc[r] + rs;
#pragma unroll
            for (int j = 0; j < 4; ++j) acc[j][r] *= fsc[r];
        }

        // ---- P (C/D layout) -> wave-private LDS -> A-fragment ------------
        {
            const int g8 = (lane >> 4) * 8;
            const int n  = lane & 15;
#pragma unroll
            for (int r = 0; r < 8; ++r) {
                pw[(g8 + r) * KVSTEP + n]      = (__bf16)s0[r];
                pw[(g8 + r) * KVSTEP + 16 + n] = (__bf16)s1[r];
            }
        }
        asm volatile("s_wait_dscnt 0" ::: "memory");   // wave-private exchange
        v16bf pa = load_a_bf16(pw, KVSTEP);

        // ---- acc += P (16x32) x V (32x64), V from LDS --------------------
#pragma unroll
        for (int j = 0; j < 4; ++j) {
            v16bf vb = load_b_bf16(&vs[cur][j * 16], HDIM);
            acc[j] = WMMA_BF16(pa, vb, acc[j]);
        }

        // staging of (cur^1) must be complete; all waves done reading cur
        wait_async_copies();
        __syncthreads();
    }

    // ---- epilogue: 1/l normalize, apply style gate, store bf16 -----------
    const int g = lane >> 4;
    const int n = lane & 15;
    float gv[4];
#pragma unroll
    for (int j = 0; j < 4; ++j) gv[j] = gate[b * DIM + h * HDIM + j * 16 + n];
    float inv[8];
#pragma unroll
    for (int r = 0; r < 8; ++r) inv[r] = 1.0f / lrow[r];
#pragma unroll
    for (int j = 0; j < 4; ++j) {
#pragma unroll
        for (int r = 0; r < 8; ++r) {
            const int seq = q0 + g * 8 + r;
            ao[(size_t)(b * SEQ + seq) * DIM + h * HDIM + j * 16 + n] =
                (__bf16)(acc[j][r] * inv[r] * gv[j]);
        }
    }
}

// ---------------------------------------------------------------------------
// Kernel 4: final projection  out = gated_attn(bf16) @ Wo(bf16) + bo -> fp32
// Each wave: 32x64 output tile.
// ---------------------------------------------------------------------------
__global__ __launch_bounds__(256) void out_proj_kernel(
    const __bf16* __restrict__ ao, const __bf16* __restrict__ Wo,
    const float* __restrict__ bo, float* __restrict__ out)
{
    const int wave = threadIdx.x >> 5;
    const int lane = threadIdx.x & 31;
    const int row0 = blockIdx.x * 256 + wave * 32;
    const int col0 = blockIdx.y * 64;

    v8f acc[2][4] = {};
    const __bf16* ap = ao + (size_t)row0 * DIM;
#pragma unroll 2
    for (int k0 = 0; k0 < DIM; k0 += 32) {
        v16bf a0 = load_a_bf16(ap + k0, DIM);
        v16bf a1 = load_a_bf16(ap + (size_t)16 * DIM + k0, DIM);
#pragma unroll
        for (int j = 0; j < 4; ++j) {
            v16bf bfr = load_b_bf16(Wo + (size_t)k0 * DIM + col0 + j * 16, DIM);
            acc[0][j] = WMMA_BF16(a0, bfr, acc[0][j]);
            acc[1][j] = WMMA_BF16(a1, bfr, acc[1][j]);
        }
    }

    const int g = lane >> 4;
    const int n = lane & 15;
#pragma unroll
    for (int m = 0; m < 2; ++m) {
#pragma unroll
        for (int j = 0; j < 4; ++j) {
            const int c  = col0 + j * 16 + n;
            const float bb = bo[c];
#pragma unroll
            for (int r = 0; r < 8; ++r)
                out[(size_t)(row0 + m * 16 + g * 8 + r) * DIM + c] = acc[m][j][r] + bb;
        }
    }
}

// ---------------------------------------------------------------------------
extern "C" void kernel_launch(void* const* d_in, const int* in_sizes, int n_in,
                              void* d_out, int out_size, void* d_ws, size_t ws_size,
                              hipStream_t stream) {
    const float* x   = (const float*)d_in[0];
    const float* sty = (const float*)d_in[1];
    const float* Wq  = (const float*)d_in[2];  const float* bq  = (const float*)d_in[3];
    const float* Wk  = (const float*)d_in[4];  const float* bk  = (const float*)d_in[5];
    const float* Wv  = (const float*)d_in[6];  const float* bv  = (const float*)d_in[7];
    const float* Ws1 = (const float*)d_in[8];  const float* bs1 = (const float*)d_in[9];
    const float* Ws2 = (const float*)d_in[10]; const float* bs2 = (const float*)d_in[11];
    const float* Wo  = (const float*)d_in[12]; const float* bo  = (const float*)d_in[13];
    float* out = (float*)d_out;

    // workspace layout (all fully overwritten every call -> deterministic):
    //   q | kT | v | gated-attn-out | x_bf16 | Wq_b | Wk_b | Wv_b | Wo_b | gate
    const size_t nElems = (size_t)BATCH * SEQ * DIM;   // 4M
    const size_t wElems = (size_t)DIM * DIM;           // 256K
    __bf16* qb  = (__bf16*)d_ws;
    __bf16* kTb = qb  + nElems;
    __bf16* vb  = kTb + nElems;
    __bf16* aob = vb  + nElems;
    __bf16* xbb = aob + nElems;
    __bf16* Wqb = xbb + nElems;
    __bf16* Wkb = Wqb + wElems;
    __bf16* Wvb = Wkb + wElems;
    __bf16* Wob = Wvb + wElems;
    float*  gtb = (float*)(Wob + wElems);

    // one-time (per call) bf16 conversions: x and the four GEMM weights
    cvt_f32_bf16_kernel<<<dim3(nElems / (256 * 8)), dim3(256), 0, stream>>>(x,  xbb);
    cvt_f32_bf16_kernel<<<dim3(wElems / (256 * 8)), dim3(256), 0, stream>>>(Wq, Wqb);
    cvt_f32_bf16_kernel<<<dim3(wElems / (256 * 8)), dim3(256), 0, stream>>>(Wk, Wkb);
    cvt_f32_bf16_kernel<<<dim3(wElems / (256 * 8)), dim3(256), 0, stream>>>(Wv, Wvb);
    cvt_f32_bf16_kernel<<<dim3(wElems / (256 * 8)), dim3(256), 0, stream>>>(Wo, Wob);

    gate_mlp_kernel<<<dim3(BATCH), dim3(DIM), 0, stream>>>(sty, Ws1, bs1, Ws2, bs2, gtb);

    qkv_proj_kernel<<<dim3(MROWS / 256, DIM / 64, 3), dim3(256), 0, stream>>>(
        xbb, Wqb, bq, Wkb, bk, Wvb, bv, qb, kTb, vb);
    flash_attn_kernel<<<dim3(SEQ / 128, NHEADS, BATCH), dim3(256), 0, stream>>>(
        qb, kTb, vb, gtb, aob);
    out_proj_kernel<<<dim3(MROWS / 256, DIM / 64), dim3(256), 0, stream>>>(
        aob, Wob, bo, out);
}
